// ExpertPool_32366873543107
// MI455X (gfx1250) — matrix operand order repaired and checked by reading, not compiled, not run on hardware
//
#include <hip/hip_runtime.h>
#include <hip/hip_bf16.h>

typedef __attribute__((ext_vector_type(16))) _Float16 v16h;
typedef __attribute__((ext_vector_type(8)))  _Float16 v8h;
typedef __attribute__((ext_vector_type(8)))  float    v8f;

#define D_MODEL   768
#define D_EXPERT  3072
#define N_EXPERTS 8
#define TOKENS    4096
#define NSLOTS    8192
#define CAP       8192
#define CHUNK     512
#define MTILE     32
#define XS        (D_MODEL + 8)   // padded LDS row stride (f16) -> conflict-free b128
#define HS        (CHUNK + 8)

#define XS_BYTES  (MTILE * XS * 2)          // 49,664
#define HS_BYTES  (MTILE * HS * 2)          // 33,280
#define SMEM_BYTES (XS_BYTES + HS_BYTES + MTILE * 4 * 3 + 16)

union ABFrag { v16h v; v8h h[2]; };

// ---------------- router: build per-expert slot lists ----------------
__global__ void moe_router(const int* __restrict__ eidx,
                           const float* __restrict__ rw,
                           int* __restrict__ counts,
                           int* __restrict__ slots,
                           float* __restrict__ wlist) {
    int s = blockIdx.x * blockDim.x + threadIdx.x;
    if (s >= NSLOTS) return;
    int e = eidx[s];
    int pos = atomicAdd(&counts[e], 1);
    slots[e * CAP + pos] = s;
    wlist[e * CAP + pos] = rw[s];
}

// ------------- transpose + fp32 -> f16 convert (per expert z) -------------
// in: (R, C) fp32 row-major   out: (C, R) f16 row-major
__global__ __launch_bounds__(256)
void transpose_cvt(const float* __restrict__ in, _Float16* __restrict__ out,
                   int R, int C) {
    __shared__ float tile[32][33];
    const size_t mat = (size_t)R * C;
    const float* ip = in + (size_t)blockIdx.z * mat;
    _Float16*    op = out + (size_t)blockIdx.z * mat;
    int c0 = blockIdx.x * 32, r0 = blockIdx.y * 32;
    #pragma unroll
    for (int i = threadIdx.y; i < 32; i += 8) {
        tile[i][threadIdx.x] = ip[(size_t)(r0 + i) * C + (c0 + threadIdx.x)];
    }
    __syncthreads();
    #pragma unroll
    for (int i = threadIdx.y; i < 32; i += 8) {
        op[(size_t)(c0 + i) * R + (r0 + threadIdx.x)] =
            (_Float16)tile[threadIdx.x][i];
    }
}

// ---------------- fused expert GEMM: silu(x Wg) * (x Wu) @ Wd ----------------
// One workgroup = (expert, 32-token tile). 256 threads = 8 waves.
__global__ __launch_bounds__(256)
void moe_wmma(const float* __restrict__ x,
              const int* __restrict__ counts,
              const int* __restrict__ slots,
              const float* __restrict__ wlist,
              const _Float16* __restrict__ wg,   // (E, D_EXPERT, D_MODEL) f16 (W^T)
              const _Float16* __restrict__ wu,   // (E, D_EXPERT, D_MODEL) f16
              const _Float16* __restrict__ wd,   // (E, D_MODEL, D_EXPERT) f16 (W^T)
              float* __restrict__ buf)           // (NSLOTS, D_MODEL)
{
    extern __shared__ char smem[];
    _Float16* xs  = (_Float16*)smem;                         // [MTILE][XS]
    _Float16* hs  = (_Float16*)(smem + XS_BYTES);            // [MTILE][HS]
    int*      ts  = (int*)  (smem + XS_BYTES + HS_BYTES);    // [MTILE]
    int*      ss  = ts + MTILE;                              // [MTILE]
    float*    wsm = (float*)(ss + MTILE);                    // [MTILE]
    int*      info = (int*)(wsm + MTILE);                    // [3]

    const int tid = threadIdx.x;

    if (tid == 0) {
        int cum = 0, e = -1, tstart = 0, cnt = 0;
        int bid = blockIdx.x;
        #pragma unroll
        for (int i = 0; i < N_EXPERTS; ++i) {
            int ce = counts[i];
            int nt = (ce + MTILE - 1) >> 5;
            if (e < 0 && bid < cum + nt) { e = i; tstart = (bid - cum) * MTILE; cnt = ce; }
            cum += nt;
        }
        info[0] = e; info[1] = tstart; info[2] = cnt;
    }
    __syncthreads();
    const int e = info[0];
    if (e < 0) return;                   // uniform across block
    const int tstart = info[1];
    const int cnt    = info[2];

    if (tid < MTILE) {
        int p = tstart + tid;
        int slot = 0, tok = 0; float w = 0.f;
        if (p < cnt) {
            slot = slots[e * CAP + p];
            w    = wlist[e * CAP + p];
            tok  = slot >> 1;            // slot = token*2 + k
        }
        ts[tid] = tok; ss[tid] = slot; wsm[tid] = w;
    }
    __syncthreads();

    // gather 32 token rows of x into LDS as f16
    for (int i = tid; i < MTILE * D_MODEL; i += 256) {
        int r = i / D_MODEL, c = i - r * D_MODEL;
        xs[r * XS + c] = (_Float16)x[(size_t)ts[r] * D_MODEL + c];
    }
    __syncthreads();

    const int wave = tid >> 5;
    const int lane = tid & 31;
    const int ln   = lane & 15;          // B column / A row within subtile
    const int hi   = lane >> 4;          // lane-half -> K sub-block

    v8f dacc[12];                        // [out-tile 0..5][M-subtile 0..1]
    #pragma unroll
    for (int i = 0; i < 12; ++i)
        dacc[i] = (v8f){0.f,0.f,0.f,0.f,0.f,0.f,0.f,0.f};

    const _Float16* wgE = wg + (size_t)e * D_EXPERT * D_MODEL;
    const _Float16* wuE = wu + (size_t)e * D_EXPERT * D_MODEL;
    const _Float16* wdE = wd + (size_t)e * D_MODEL * D_EXPERT;

    for (int c = 0; c < D_EXPERT / CHUNK; ++c) {
        // ---- gate/up GEMMs for this De chunk, silu*up -> hs ----
        for (int t4 = 0; t4 < 4; ++t4) {
            int ncl = wave * 64 + t4 * 16;           // col within chunk
            int n   = c * CHUNK + ncl + ln;          // global De col for this lane
            v8f g0 = (v8f){0.f,0.f,0.f,0.f,0.f,0.f,0.f,0.f};
            v8f g1 = g0, u0 = g0, u1 = g0;
            const _Float16* wgp = wgE + (size_t)n * D_MODEL + hi * 8;
            const _Float16* wup = wuE + (size_t)n * D_MODEL + hi * 8;
            #pragma unroll
            for (int k0 = 0; k0 < D_MODEL; k0 += 32) {
                ABFrag a0, a1, bg, bu;
                int ka = k0 + hi * 8;
                a0.h[0] = *(const v8h*)&xs[ln * XS + ka];
                a0.h[1] = *(const v8h*)&xs[ln * XS + ka + 16];
                a1.h[0] = *(const v8h*)&xs[(16 + ln) * XS + ka];
                a1.h[1] = *(const v8h*)&xs[(16 + ln) * XS + ka + 16];
                bg.h[0] = *(const v8h*)(wgp + k0);
                bg.h[1] = *(const v8h*)(wgp + k0 + 16);
                bu.h[0] = *(const v8h*)(wup + k0);
                bu.h[1] = *(const v8h*)(wup + k0 + 16);
                g0 = __builtin_amdgcn_wmma_f32_16x16x32_f16(false, a0.v, false, bg.v,
                                                            (short)0, g0, false, false);
                g1 = __builtin_amdgcn_wmma_f32_16x16x32_f16(false, a1.v, false, bg.v,
                                                            (short)0, g1, false, false);
                u0 = __builtin_amdgcn_wmma_f32_16x16x32_f16(false, a0.v, false, bu.v,
                                                            (short)0, u0, false, false);
                u1 = __builtin_amdgcn_wmma_f32_16x16x32_f16(false, a1.v, false, bu.v,
                                                            (short)0, u1, false, false);
            }
            #pragma unroll
            for (int j = 0; j < 8; ++j) {
                float gv0 = g0[j];
                float hv0 = (gv0 / (1.f + __expf(-gv0))) * u0[j];
                hs[(j + 8 * hi) * HS + ncl + ln] = (_Float16)hv0;
                float gv1 = g1[j];
                float hv1 = (gv1 / (1.f + __expf(-gv1))) * u1[j];
                hs[(16 + j + 8 * hi) * HS + ncl + ln] = (_Float16)hv1;
            }
        }
        __syncthreads();
        // ---- down GEMM partial over this chunk ----
        #pragma unroll
        for (int ot = 0; ot < 6; ++ot) {
            int n = (wave * 6 + ot) * 16 + ln;       // D_MODEL output col
            const _Float16* wdp = wdE + (size_t)n * D_EXPERT + c * CHUNK + hi * 8;
            v8f acc0 = dacc[ot * 2];
            v8f acc1 = dacc[ot * 2 + 1];
            #pragma unroll
            for (int ko = 0; ko < CHUNK; ko += 32) {
                ABFrag a0, a1, b;
                int ka = ko + hi * 8;
                a0.h[0] = *(const v8h*)&hs[ln * HS + ka];
                a0.h[1] = *(const v8h*)&hs[ln * HS + ka + 16];
                a1.h[0] = *(const v8h*)&hs[(16 + ln) * HS + ka];
                a1.h[1] = *(const v8h*)&hs[(16 + ln) * HS + ka + 16];
                b.h[0]  = *(const v8h*)(wdp + ko);
                b.h[1]  = *(const v8h*)(wdp + ko + 16);
                acc0 = __builtin_amdgcn_wmma_f32_16x16x32_f16(false, a0.v, false, b.v,
                                                              (short)0, acc0, false, false);
                acc1 = __builtin_amdgcn_wmma_f32_16x16x32_f16(false, a1.v, false, b.v,
                                                              (short)0, acc1, false, false);
            }
            dacc[ot * 2]     = acc0;
            dacc[ot * 2 + 1] = acc1;
        }
        __syncthreads();
    }

    // ---- scale by routing weight, scatter to per-slot buffer ----
    #pragma unroll
    for (int ot = 0; ot < 6; ++ot) {
        int n = (wave * 6 + ot) * 16 + ln;
        #pragma unroll
        for (int j = 0; j < 8; ++j) {
            int m0 = j + 8 * hi;
            if (tstart + m0 < cnt) {
                buf[(size_t)ss[m0] * D_MODEL + n] = dacc[ot * 2][j] * wsm[m0];
            }
            int m1 = 16 + j + 8 * hi;
            if (tstart + m1 < cnt) {
                buf[(size_t)ss[m1] * D_MODEL + n] = dacc[ot * 2 + 1][j] * wsm[m1];
            }
        }
    }
}

// ---------------- out[t] = buf[2t] + buf[2t+1]  (float4) ----------------
__global__ void moe_reduce(const float4* __restrict__ buf, float4* __restrict__ out) {
    int i = blockIdx.x * blockDim.x + threadIdx.x;
    if (i >= TOKENS * D_MODEL / 4) return;
    const int DQ = D_MODEL / 4;
    int t = i / DQ, d = i - t * DQ;
    float4 a = buf[(size_t)(2 * t) * DQ + d];
    float4 b = buf[(size_t)(2 * t + 1) * DQ + d];
    out[i] = make_float4(a.x + b.x, a.y + b.y, a.z + b.z, a.w + b.w);
}

extern "C" void kernel_launch(void* const* d_in, const int* in_sizes, int n_in,
                              void* d_out, int out_size, void* d_ws, size_t ws_size,
                              hipStream_t stream) {
    const float* x     = (const float*)d_in[0];
    const float* rw    = (const float*)d_in[1];
    const float* wgate = (const float*)d_in[2];
    const float* wup   = (const float*)d_in[3];
    const float* wdown = (const float*)d_in[4];
    const int*   eidx  = (const int*)d_in[5];
    float* out = (float*)d_out;

    const size_t MATB = (size_t)N_EXPERTS * D_MODEL * D_EXPERT * sizeof(_Float16); // 37.75 MB
    char* ws = (char*)d_ws;
    _Float16* wg_t   = (_Float16*)(ws);                       // (E, De, D) f16
    _Float16* wu_t   = (_Float16*)(ws + MATB);                // (E, De, D) f16
    _Float16* wd_t   = (_Float16*)(ws + 2 * MATB);            // (E, D, De) f16
    float*    buf    = (float*)   (ws + 3 * MATB);            // (NSLOTS, D) f32
    size_t bufB = (size_t)NSLOTS * D_MODEL * sizeof(float);   // 25.2 MB
    int*      counts = (int*)     (ws + 3 * MATB + bufB);
    int*      slots  = (int*)     (ws + 3 * MATB + bufB + 32);
    float*    wlist  = (float*)   (ws + 3 * MATB + bufB + 32 +
                                   (size_t)N_EXPERTS * CAP * sizeof(int));

    hipMemsetAsync(counts, 0, N_EXPERTS * sizeof(int), stream);

    // weight transpose + f16 convert (coalesced LDS-tiled)
    {
        dim3 blk(32, 8);
        dim3 g1(D_EXPERT / 32, D_MODEL / 32, N_EXPERTS);   // gate/up: (768,3072)->(3072,768)
        transpose_cvt<<<g1, blk, 0, stream>>>(wgate, wg_t, D_MODEL, D_EXPERT);
        transpose_cvt<<<g1, blk, 0, stream>>>(wup,   wu_t, D_MODEL, D_EXPERT);
        dim3 g2(D_MODEL / 32, D_EXPERT / 32, N_EXPERTS);   // down: (3072,768)->(768,3072)
        transpose_cvt<<<g2, blk, 0, stream>>>(wdown, wd_t, D_EXPERT, D_MODEL);
    }

    moe_router<<<NSLOTS / 256, 256, 0, stream>>>(eidx, rw, counts, slots, wlist);

    // max tiles = NSLOTS/32 + N_EXPERTS (padding) = 264
    moe_wmma<<<NSLOTS / MTILE + N_EXPERTS, 256, SMEM_BYTES, stream>>>(
        x, counts, slots, wlist, wg_t, wu_t, wd_t, buf);

    moe_reduce<<<(TOKENS * D_MODEL / 4 + 255) / 256, 256, 0, stream>>>(
        (const float4*)buf, (float4*)out);
}
